// NeuralClustering_2456721293669
// MI455X (gfx1250) — matrix operand order repaired and checked by reading, not compiled, not run on hardware
//
#include <hip/hip_runtime.h>
#include <hip/hip_bf16.h>

typedef __attribute__((ext_vector_type(16))) _Float16 v16h;
typedef __attribute__((ext_vector_type(8)))  float    v8f;

#define WMMA_F16(a,b,c) __builtin_amdgcn_wmma_f32_16x16x32_f16(false,(a),false,(b),(short)0,(c),false,false)

// ---------------- packed-weight layout (f16 element offsets) ----------------
// unit = one (nTile,kChunk) B-fragment = 32 lanes * 16 f16 = 512 elements
static constexpr size_t OFF_H0 = 0;        // 8  units (K=32 pad, N=128)
static constexpr size_t OFF_H1 = 4096;     // 32 units (128x128)
static constexpr size_t OFF_H2 = 20480;
static constexpr size_t OFF_H3 = 36864;
static constexpr size_t OFF_H4 = 53248;    // 64 units (128x256)
static constexpr size_t OFF_G0 = 86016;    // 64 units (256x128)
static constexpr size_t OFF_G1 = 118784;
static constexpr size_t OFF_G2 = 135168;
static constexpr size_t OFF_G3 = 151552;
static constexpr size_t OFF_G4 = 167936;
static constexpr size_t OFF_G5 = 184320;   // 128 units (128x512)
static constexpr size_t OFF_E0 = 249856;   // 192 units (768x128)
static constexpr size_t OFF_E1 = 348160;
static constexpr size_t OFF_E2 = 364544;
static constexpr size_t OFF_E3 = 380928;
static constexpr size_t OFF_E4 = 397312;
static constexpr size_t OFF_E5 = 413696;   // 4 units (128x16pad, out=1)
static constexpr size_t PACK_TOTAL = 415744;

// ---------------- workspace layout (bytes) ----------------
static constexpr size_t WS_PACK = 0;                       // f16[415744]
static constexpr size_t WS_HK   = 831488;                  // f32 [256][8][256]
static constexpr size_t WS_HN   = WS_HK  + 2097152;        // f32 [256][256]
static constexpr size_t WS_Q    = WS_HN  + 262144;         // f32 [256][256]
static constexpr size_t WS_HALL = WS_Q   + 262144;         // f32 [256][17][256]
static constexpr size_t WS_GS   = WS_HALL+ 4456448;        // f32 [256][17][512]
static constexpr size_t WS_UU   = WS_GS  + 8912896;        // f32 [256][9][768]
static constexpr size_t WS_CNT  = WS_UU  + 7077888;        // i32 [256][8]
static constexpr size_t WS_KS   = WS_CNT + 8192;           // i32 [256]

// ---------------- fragment-layout helpers (CDNA5 16-bit WMMA layouts) -------
static __device__ __forceinline__ int aK32(int half, int e){
  // 16-bit A 16x32: lane = half*16+m ; element e -> K within 32-chunk
  return (e < 8) ? (half*8 + e) : (8 + half*8 + e);
}

static __device__ __forceinline__ v16h gather_a(const float* src, int ld, int kBase, int kLimit){
  const int lane = (int)(threadIdx.x & 31);
  const int half = lane >> 4;
  const int m    = lane & 15;
  v16h a;
#pragma unroll
  for (int e = 0; e < 16; ++e){
    int k = kBase + aK32(half, e);
    float v = (k < kLimit) ? src[(size_t)m * ld + k] : 0.0f;
    a[e] = (_Float16)v;
  }
  return a;
}

// layer0: A gathered from row-major f32 global, out dim fixed 128 (8 nTiles)
template<int NKC>
static __device__ __forceinline__ void layer0_from_global(
    const float* src, int ld, int kLimit,
    const _Float16* pack0, const float* bias, float slope, v8f out[8])
{
  const int lane = (int)(threadIdx.x & 31);
  const v8f vzero = {};
#pragma unroll
  for (int nt = 0; nt < 8; ++nt) out[nt] = vzero;
  for (int kc = 0; kc < NKC; ++kc){
    v16h a = gather_a(src, ld, kc*32, kLimit);
#pragma unroll
    for (int nt = 0; nt < 8; ++nt){
      v16h w = ((const v16h*)(pack0 + (size_t)(nt*NKC + kc)*512))[lane];
      out[nt] = WMMA_F16(a, w, out[nt]);
    }
  }
  const int c16 = lane & 15;
#pragma unroll
  for (int nt = 0; nt < 8; ++nt){
    float bb = bias[nt*16 + c16];
#pragma unroll
    for (int r = 0; r < 8; ++r){
      float x = out[nt][r] + bb;
      out[nt][r] = (x > 0.0f) ? x : slope * x;
    }
  }
}

// hidden layer 128->128 reading A fragments from per-wave LDS stage
static __device__ __forceinline__ void hidden_layer(
    const _Float16* aStage, const _Float16* pack, const float* bias, float slope, v8f out[8])
{
  const int lane = (int)(threadIdx.x & 31);
  const v16h* aS = (const v16h*)aStage;
#pragma unroll
  for (int nt = 0; nt < 8; ++nt){
    v8f c = {};
#pragma unroll
    for (int kc = 0; kc < 4; ++kc){
      v16h a = aS[kc*32 + lane];
      v16h w = ((const v16h*)(pack + (size_t)(nt*4 + kc)*512))[lane];
      c = WMMA_F16(a, w, c);
    }
    float bb = bias[nt*16 + (lane & 15)];
#pragma unroll
    for (int r = 0; r < 8; ++r){
      float x = c[r] + bb;
      c[r] = (x > 0.0f) ? x : slope * x;
    }
    out[nt] = c;
  }
}

// write 16x128 C tiles into per-wave LDS stage in A-fragment order for next layer
static __device__ __forceinline__ void store_stage(_Float16* aStage, const v8f out[8]){
  const int lane  = (int)(threadIdx.x & 31);
  const int halfC = lane >> 4;
  const int c16   = lane & 15;
#pragma unroll
  for (int nt = 0; nt < 8; ++nt){
    int col = nt*16 + c16;           // becomes K index of next layer
    int kc  = col >> 5, k32 = col & 31;
    int halfA, e;
    if (k32 < 16){ halfA = k32 >> 3; e = k32 & 7; }
    else         { int t = k32 - 16; halfA = t >> 3; e = (t & 7) + 8; }
#pragma unroll
    for (int r = 0; r < 8; ++r){
      int m = r + 8*halfC;           // C layout: M = r + 8*half
      aStage[(size_t)((kc*32 + (halfA*16 + m))*16 + e)] = (_Float16)out[nt][r];
    }
  }
}

// ---------------- setup kernels ----------------
__global__ void zero_f32(float* __restrict__ p, int nElem){
  int id = (int)(blockIdx.x*256u + threadIdx.x);
  if (id < nElem) p[id] = 0.0f;
}

// pack W[in][out] (f32) -> B-fragment f16: [nt][kc][lane][e], K = e<8 ? half*8 + e : 16 + half*8 + (e-8)
__global__ void pack_weights(const float* __restrict__ W, int inDim, int outDim,
                             int Kp, int Np, _Float16* __restrict__ dst){
  int id = (int)(blockIdx.x*256u + threadIdx.x);
  int total = (Np/16)*(Kp/32)*512;
  if (id >= total) return;
  int e    = id & 15;
  int lane = (id >> 4) & 31;
  int unit = id >> 9;
  int nKc = Kp/32;
  int nt = unit / nKc, kc = unit % nKc;
  int half = lane >> 4;
  int ncol = nt*16 + (lane & 15);
  int k = kc*32 + ((e < 8) ? (half*8 + e) : (16 + half*8 + (e - 8)));
  float v = 0.0f;
  if (k < inDim && ncol < outDim) v = W[(size_t)k*outDim + ncol];
  dst[id] = (_Float16)v;
}

__global__ void stats_kernel(const int* __restrict__ cs, const int* __restrict__ nPtr,
                             int* __restrict__ counts, int* __restrict__ Ks){
  __shared__ int cnt[8];
  __shared__ int mx;
  const int b = (int)blockIdx.x;
  const int tid = (int)threadIdx.x;
  if (tid < 8) cnt[tid] = 0;
  if (tid == 0) mx = -1;
  __syncthreads();
  const int n = nPtr[0];
  for (int t = tid; t < n; t += 256){
    int c = cs[(size_t)b*800 + t];
    atomicAdd(&cnt[c & 7], 1);
    atomicMax(&mx, c);
  }
  __syncthreads();
  if (tid < 8) counts[b*8 + tid] = cnt[tid];
  if (tid == 0) Ks[b] = mx;
}

// ---------------- kernel 1: fused h-MLP + segment reductions ----------------
__global__ __launch_bounds__(256) void h_kernel(
  const float* __restrict__ data, const int* __restrict__ cs, const int* __restrict__ nPtr,
  const _Float16* __restrict__ pack,
  const float* b0, const float* b1, const float* b2, const float* b3, const float* b4,
  const float* a0, const float* a1, const float* a2, const float* a3,
  float* __restrict__ HkG, float* __restrict__ hnG, float* __restrict__ QG)
{
  __shared__ __align__(32) _Float16 stage[8][2048];  // per-wave A-stage (K<=128)
  __shared__ float accS[2560];                       // Hk[8][256] | hn[256] | Q[256]
  const int b = (int)blockIdx.x / 7;
  const int chunk = (int)blockIdx.x % 7;
  const int tid = (int)threadIdx.x, wave = tid >> 5, lane = tid & 31;
  for (int i = tid; i < 2560; i += 256) accS[i] = 0.0f;
  __syncthreads();
  const int n = nPtr[0];
  const int base = chunk*128 + wave*16;
  if (base < 800){  // wave-uniform: EXEC stays all-ones inside
    v8f out[8];
    layer0_from_global<1>(data + ((size_t)b*800 + base)*2, 2, 2, pack + OFF_H0, b0, a0[0], out);
    store_stage(stage[wave], out);
    hidden_layer(stage[wave], pack + OFF_H1, b1, a1[0], out); store_stage(stage[wave], out);
    hidden_layer(stage[wave], pack + OFF_H2, b2, a2[0], out); store_stage(stage[wave], out);
    hidden_layer(stage[wave], pack + OFF_H3, b3, a3[0], out); store_stage(stage[wave], out);
    // final 128->256, stream nTiles straight into LDS reduction (ds_add_f32)
    const v16h* aS = (const v16h*)stage[wave];
    const int halfC = lane >> 4;
    int dst[8]; int hnFlag[8];
#pragma unroll
    for (int r = 0; r < 8; ++r){
      int t = base + r + 8*halfC;
      dst[r]    = (t < n) ? (cs[(size_t)b*800 + t] & 7) * 256 : 2304;  // Hk[c] or Q
      hnFlag[r] = (t == n);
    }
    for (int nt = 0; nt < 16; ++nt){
      v8f c = {};
#pragma unroll
      for (int kc = 0; kc < 4; ++kc){
        v16h a = aS[kc*32 + lane];
        v16h w = ((const v16h*)(pack + OFF_H4 + (size_t)(nt*4 + kc)*512))[lane];
        c = WMMA_F16(a, w, c);
      }
      int col = nt*16 + (lane & 15);
      float bb = b4[col];  // final layer: bias, no activation
#pragma unroll
      for (int r = 0; r < 8; ++r){
        float v = c[r] + bb;
        atomicAdd(&accS[dst[r] + col], v);
        if (hnFlag[r]) atomicAdd(&accS[2048 + col], v);
      }
    }
  }
  __syncthreads();
  for (int i = tid; i < 2048; i += 256) atomicAdd(&HkG[(size_t)b*2048 + i], accS[i]);
  atomicAdd(&hnG[(size_t)b*256 + tid], accS[2048 + tid]);
  atomicAdd(&QG [(size_t)b*256 + tid], accS[2304 + tid]);
}

// ---------------- kernel 2: build Hall = [Hk, Hk+hn, hn] ----------------
__global__ void hall_build(const float* __restrict__ Hk, const float* __restrict__ hn,
                           float* __restrict__ Hall){
  int id = (int)(blockIdx.x*256u + threadIdx.x);   // B*17*256 exactly
  int c = id & 255;
  int j = (id >> 8) % 17;
  int b = id / (17*256);
  float v;
  if (j < 8)       v = Hk[((size_t)b*8 + j)*256 + c];
  else if (j < 16) v = Hk[((size_t)b*8 + (j-8))*256 + c] + hn[(size_t)b*256 + c];
  else             v = hn[(size_t)b*256 + c];
  Hall[id] = v;
}

// ---------------- kernel 3: g-MLP (4352 rows, 256->...->512, masked) --------
__global__ __launch_bounds__(256) void g_kernel(
  const float* __restrict__ Hall, const _Float16* __restrict__ pack,
  const float* b0, const float* b1, const float* b2, const float* b3, const float* b4, const float* b5,
  const float* a0, const float* a1, const float* a2, const float* a3, const float* a4,
  const int* __restrict__ counts, float* __restrict__ gs)
{
  __shared__ __align__(32) _Float16 stage[8][2048];
  const int tid = (int)threadIdx.x, wave = tid >> 5, lane = tid & 31;
  const int rowBase = ((int)blockIdx.x * 8 + wave) * 16;
  v8f out[8];
  layer0_from_global<8>(Hall + (size_t)rowBase*256, 256, 256, pack + OFF_G0, b0, a0[0], out);
  store_stage(stage[wave], out);
  hidden_layer(stage[wave], pack + OFF_G1, b1, a1[0], out); store_stage(stage[wave], out);
  hidden_layer(stage[wave], pack + OFF_G2, b2, a2[0], out); store_stage(stage[wave], out);
  hidden_layer(stage[wave], pack + OFF_G3, b3, a3[0], out); store_stage(stage[wave], out);
  hidden_layer(stage[wave], pack + OFF_G4, b4, a4[0], out); store_stage(stage[wave], out);
  const v16h* aS = (const v16h*)stage[wave];
  const int halfC = lane >> 4;
  for (int nt = 0; nt < 32; ++nt){
    v8f c = {};
#pragma unroll
    for (int kc = 0; kc < 4; ++kc){
      v16h a = aS[kc*32 + lane];
      v16h w = ((const v16h*)(pack + OFF_G5 + (size_t)(nt*4 + kc)*512))[lane];
      c = WMMA_F16(a, w, c);
    }
    int col = nt*16 + (lane & 15);
    float bb = b5[col];
#pragma unroll
    for (int r = 0; r < 8; ++r){
      int row = rowBase + r + 8*halfC;
      int bi = row / 17, j = row - bi*17;
      float mk = (j == 16) ? 1.0f : ((counts[bi*8 + (j & 7)] > 0) ? 1.0f : 0.0f);
      gs[(size_t)row*512 + col] = (c[r] + bb) * mk;
    }
  }
}

// ---------------- kernel 4: assemble G / uu / G_mask ----------------
__global__ __launch_bounds__(256) void assemble_kernel(
  const float* __restrict__ gs, const float* __restrict__ QG,
  const int* __restrict__ counts, const int* __restrict__ Ks,
  float* __restrict__ uu, float* __restrict__ maskOut)
{
  const int b = (int)blockIdx.x;
  const int tid = (int)threadIdx.x;
  const int ks = Ks[b];
  const bool mv = (ks < 7);  // need_move: Ks < K-1
  for (int col = tid; col < 512; col += 256){
    float vk[8]; float sum = 0.0f;
#pragma unroll
    for (int k = 0; k < 8; ++k){ vk[k] = gs[((size_t)b*17 + k)*512 + col]; sum += vk[k]; }
    float GK = sum + gs[((size_t)b*17 + 16)*512 + col];
#pragma unroll
    for (int k = 0; k < 8; ++k){
      float mk = (counts[b*8 + k] > 0) ? 1.0f : 0.0f;
      float val = (sum - vk[k] + gs[((size_t)b*17 + 8 + k)*512 + col]) * mk;
      if (mv && (k == ks + 1)) val = GK;   // move GK into the "new cluster" slot
      uu[((size_t)b*9 + k)*768 + col] = val;
    }
    uu[((size_t)b*9 + 8)*768 + col] = mv ? 0.0f : GK;
  }
  // broadcast Q into cols [512,768)
  {
    float q = QG[(size_t)b*256 + tid];
#pragma unroll
    for (int row = 0; row < 9; ++row)
      uu[((size_t)b*9 + row)*768 + 512 + tid] = q;
  }
  if (tid < 9) maskOut[b*9 + tid] = (tid <= ks + 1) ? 1.0f : 0.0f;
}

// ---------------- kernel 5: E-MLP (2304 rows, 768->...->1) ----------------
__global__ __launch_bounds__(256) void e_kernel(
  const float* __restrict__ uu, const _Float16* __restrict__ pack,
  const float* b0, const float* b1, const float* b2, const float* b3, const float* b4, const float* b5,
  const float* a0, const float* a1, const float* a2, const float* a3, const float* a4,
  float* __restrict__ Eout)
{
  __shared__ __align__(32) _Float16 stage[8][2048];
  const int tid = (int)threadIdx.x, wave = tid >> 5, lane = tid & 31;
  const int rowBase = ((int)blockIdx.x * 8 + wave) * 16;
  v8f out[8];
  layer0_from_global<24>(uu + (size_t)rowBase*768, 768, 768, pack + OFF_E0, b0, a0[0], out);
  store_stage(stage[wave], out);
  hidden_layer(stage[wave], pack + OFF_E1, b1, a1[0], out); store_stage(stage[wave], out);
  hidden_layer(stage[wave], pack + OFF_E2, b2, a2[0], out); store_stage(stage[wave], out);
  hidden_layer(stage[wave], pack + OFF_E3, b3, a3[0], out); store_stage(stage[wave], out);
  hidden_layer(stage[wave], pack + OFF_E4, b4, a4[0], out); store_stage(stage[wave], out);
  // final 128 -> 1 (N padded to 16, keep col 0)
  const v16h* aS = (const v16h*)stage[wave];
  v8f c = {};
#pragma unroll
  for (int kc = 0; kc < 4; ++kc){
    v16h a = aS[kc*32 + lane];
    v16h w = ((const v16h*)(pack + OFF_E5 + (size_t)kc*512))[lane];
    c = WMMA_F16(a, w, c);
  }
  float bb = b5[0];
  if ((lane & 15) == 0){
    const int halfC = lane >> 4;
#pragma unroll
    for (int r = 0; r < 8; ++r){
      int row = rowBase + r + 8*halfC;
      Eout[row] = c[r] + bb;
    }
  }
}

// ---------------- host ----------------
static void launch_pack(const float* W, int inD, int outD, int Kp, int Np,
                        _Float16* dst, hipStream_t s){
  int total = (Np/16)*(Kp/32)*512;
  pack_weights<<<(total + 255)/256, 256, 0, s>>>(W, inD, outD, Kp, Np, dst);
}

extern "C" void kernel_launch(void* const* d_in, const int* in_sizes, int n_in,
                              void* d_out, int out_size, void* d_ws, size_t ws_size,
                              hipStream_t stream)
{
  (void)in_sizes; (void)n_in; (void)out_size; (void)ws_size;
  // flattening: top-level dict insertion order; each params pytree in sorted-key
  // order ('W' leaves, then 'a', then 'b').
  const float* data = (const float*)d_in[0];
  const int*   cs   = (const int*)  d_in[1];
  const int*   nP   = (const int*)  d_in[2];
  const float* hW[5]; const float* hA[4]; const float* hB[5];
  for (int i = 0; i < 5; ++i) hW[i] = (const float*)d_in[3 + i];
  for (int i = 0; i < 4; ++i) hA[i] = (const float*)d_in[8 + i];
  for (int i = 0; i < 5; ++i) hB[i] = (const float*)d_in[12 + i];
  const float* gW[6]; const float* gA[5]; const float* gB[6];
  for (int i = 0; i < 6; ++i) gW[i] = (const float*)d_in[17 + i];
  for (int i = 0; i < 5; ++i) gA[i] = (const float*)d_in[23 + i];
  for (int i = 0; i < 6; ++i) gB[i] = (const float*)d_in[28 + i];
  const float* eW[6]; const float* eA[5]; const float* eB[6];
  for (int i = 0; i < 6; ++i) eW[i] = (const float*)d_in[34 + i];
  for (int i = 0; i < 5; ++i) eA[i] = (const float*)d_in[40 + i];
  for (int i = 0; i < 6; ++i) eB[i] = (const float*)d_in[45 + i];

  char* ws = (char*)d_ws;
  _Float16* pack  = (_Float16*)(ws + WS_PACK);
  float*    HkG   = (float*)(ws + WS_HK);
  float*    hnG   = (float*)(ws + WS_HN);
  float*    QG    = (float*)(ws + WS_Q);
  float*    Hall  = (float*)(ws + WS_HALL);
  float*    gsB   = (float*)(ws + WS_GS);
  float*    uu    = (float*)(ws + WS_UU);
  int*      cnt   = (int*)  (ws + WS_CNT);
  int*      Ks    = (int*)  (ws + WS_KS);
  float*    Eout  = (float*)d_out;           // [256*9] E, then [256*9] G_mask
  float*    Mout  = Eout + 2304;

  // 0) zero the accumulators (Hk|hn|Q contiguous: 655360 floats)
  zero_f32<<<2560, 256, 0, stream>>>(HkG, 655360);
  // 0b) cluster counts + Ks
  stats_kernel<<<256, 256, 0, stream>>>(cs, nP, cnt, Ks);
  // 1) pack weights into B-fragment order (f16)
  launch_pack(hW[0],   2, 128,  32, 128, pack + OFF_H0, stream);
  launch_pack(hW[1], 128, 128, 128, 128, pack + OFF_H1, stream);
  launch_pack(hW[2], 128, 128, 128, 128, pack + OFF_H2, stream);
  launch_pack(hW[3], 128, 128, 128, 128, pack + OFF_H3, stream);
  launch_pack(hW[4], 128, 256, 128, 256, pack + OFF_H4, stream);
  launch_pack(gW[0], 256, 128, 256, 128, pack + OFF_G0, stream);
  launch_pack(gW[1], 128, 128, 128, 128, pack + OFF_G1, stream);
  launch_pack(gW[2], 128, 128, 128, 128, pack + OFF_G2, stream);
  launch_pack(gW[3], 128, 128, 128, 128, pack + OFF_G3, stream);
  launch_pack(gW[4], 128, 128, 128, 128, pack + OFF_G4, stream);
  launch_pack(gW[5], 128, 512, 128, 512, pack + OFF_G5, stream);
  launch_pack(eW[0], 768, 128, 768, 128, pack + OFF_E0, stream);
  launch_pack(eW[1], 128, 128, 128, 128, pack + OFF_E1, stream);
  launch_pack(eW[2], 128, 128, 128, 128, pack + OFF_E2, stream);
  launch_pack(eW[3], 128, 128, 128, 128, pack + OFF_E3, stream);
  launch_pack(eW[4], 128, 128, 128, 128, pack + OFF_E4, stream);
  launch_pack(eW[5], 128,   1, 128,  16, pack + OFF_E5, stream);

  // 2) fused h-MLP + segment reductions  (256 batches x 7 chunks of 128 tokens)
  h_kernel<<<256*7, 256, 0, stream>>>(data, cs, nP, pack,
      hB[0], hB[1], hB[2], hB[3], hB[4], hA[0], hA[1], hA[2], hA[3],
      HkG, hnG, QG);

  // 3) Hall = [Hk, Hk+hn, hn]
  hall_build<<<4352, 256, 0, stream>>>(HkG, hnG, Hall);

  // 4) g-MLP (4352 rows = 34 groups * 8 waves * 16)
  g_kernel<<<34, 256, 0, stream>>>(Hall, pack,
      gB[0], gB[1], gB[2], gB[3], gB[4], gB[5],
      gA[0], gA[1], gA[2], gA[3], gA[4], cnt, gsB);

  // 5) assemble G / uu / G_mask
  assemble_kernel<<<256, 256, 0, stream>>>(gsB, QG, cnt, Ks, uu, Mout);

  // 6) E-MLP (2304 rows = 18 groups * 8 waves * 16)
  e_kernel<<<18, 256, 0, stream>>>(uu, pack,
      eB[0], eB[1], eB[2], eB[3], eB[4], eB[5],
      eA[0], eA[1], eA[2], eA[3], eA[4], Eout);
}
